// Site_net_13795434954994
// MI455X (gfx1250) — matrix operand order repaired and checked by reading, not compiled, not run on hardware
//
#include <hip/hip_runtime.h>
#include <math.h>

typedef __attribute__((ext_vector_type(2)))  __bf16 bf16x2;
typedef __attribute__((ext_vector_type(8)))  __bf16 v8bf;
typedef __attribute__((ext_vector_type(16))) __bf16 v16bf;
typedef __attribute__((ext_vector_type(8)))  float  v8f;

#define N_SAMP 8192
#define D 256
#define NTYPE 64
#define MROWS 64       // sample rows per tile
#define ASTRIDE 264    // bf16 elems per LDS row: 528B, 16B-aligned, optimal 2-phase banking

// d_ws int32 layout
#define WS_COUNTS  0        // [64]
#define WS_OFFSETS 64       // [64]
#define WS_SLOT    128      // [8192]
#define WS_PERM    8320     // [8192]
#define WS_TILES   16512    // [<=256] packed (type<<16)|rowStart
#define WS_NTILES  17152

__global__ void k_zero(int* ws) {
    int t = threadIdx.x;
    if (t < NTYPE) ws[WS_COUNTS + t] = 0;
}

__global__ void k_count(const int* __restrict__ bi, int* ws) {
    int i = blockIdx.x * blockDim.x + threadIdx.x;
    if (i < N_SAMP) ws[WS_SLOT + i] = atomicAdd(&ws[WS_COUNTS + bi[i]], 1);
}

__global__ void k_scan_tiles(int* ws) {
    if (threadIdx.x == 0) {
        int off = 0, nt = 0;
        for (int t = 0; t < NTYPE; ++t) {
            int c = ws[WS_COUNTS + t];
            ws[WS_OFFSETS + t] = off;
            for (int r = 0; r < c; r += MROWS) ws[WS_TILES + nt++] = (t << 16) | r;
            off += c;
        }
        ws[WS_NTILES] = nt;
    }
}

__global__ void k_scatter(const int* __restrict__ bi, int* ws) {
    int i = blockIdx.x * blockDim.x + threadIdx.x;
    if (i < N_SAMP) {
        int t = bi[i];
        ws[WS_PERM + ws[WS_OFFSETS + t] + ws[WS_SLOT + i]] = i;
    }
}

// Split a lane-contiguous run of 16 f32 into bf16 hi (RNE) + bf16 lo (residual).
__device__ __forceinline__ void split_frag(const float* __restrict__ p,
                                           v16bf& hi, v16bf& lo) {
    float f[16];
    #pragma unroll
    for (int j = 0; j < 4; ++j) {
        float4 q = reinterpret_cast<const float4*>(p)[j];
        f[j * 4 + 0] = q.x; f[j * 4 + 1] = q.y; f[j * 4 + 2] = q.z; f[j * 4 + 3] = q.w;
    }
    #pragma unroll
    for (int e = 0; e < 16; ++e) {
        __bf16 h = (__bf16)f[e];
        hi[e] = h;
        lo[e] = (__bf16)(f[e] - (float)h);
    }
}

// One block per (type, 64-row tile): 8 waves x (2 column tiles of 16) x (4 row
// sub-tiles of 16). D = A x B + C via V_WMMA_F32_16X16X32_BF16 with hi/lo split:
//   acc += Ah*Bh + Al*Bh + Ah*Bl   (~2^-17 relative error, fp32-class)
__global__ __launch_bounds__(256) void k_typed_gemm(
    const float* __restrict__ desc, const float* __restrict__ layer1,
    const float* __restrict__ W, const float* __restrict__ bias,
    const int* __restrict__ ws, float* __restrict__ out) {

    __shared__ unsigned short Ahi[MROWS * ASTRIDE];
    __shared__ unsigned short Alo[MROWS * ASTRIDE];
    __shared__ int rowIdx[MROWS];

    int nt = ws[WS_NTILES];
    if ((int)blockIdx.x >= nt) return;            // wave-uniform exit

    int pk   = ws[WS_TILES + blockIdx.x];
    int t    = pk >> 16;
    int r    = pk & 0xFFFF;
    int cnt  = ws[WS_COUNTS + t] - r;  if (cnt > MROWS) cnt = MROWS;
    int base = ws[WS_OFFSETS + t] + r;
    const int* perm = ws + WS_PERM;

    // ---- stage A: gather 64 descriptor rows, pre-split into bf16 hi/lo ----
    int tid = threadIdx.x;
    int m = tid >> 2, c = tid & 3;                // 4 threads/row, 64 floats each
    int row = (m < cnt) ? perm[base + m] : -1;
    if (c == 0) rowIdx[m] = row;
    const float* src = desc + (size_t)(row < 0 ? 0 : row) * D + c * 64;
    unsigned short* hrow = Ahi + m * ASTRIDE + c * 64;
    unsigned short* lrow = Alo + m * ASTRIDE + c * 64;
    #pragma unroll
    for (int j = 0; j < 16; ++j) {
        float4 f = make_float4(0.f, 0.f, 0.f, 0.f);
        if (row >= 0) f = reinterpret_cast<const float4*>(src)[j];
        bf16x2 h01, h23, l01, l23;
        h01.x = (__bf16)f.x; h01.y = (__bf16)f.y;
        h23.x = (__bf16)f.z; h23.y = (__bf16)f.w;
        l01.x = (__bf16)(f.x - (float)h01.x); l01.y = (__bf16)(f.y - (float)h01.y);
        l23.x = (__bf16)(f.z - (float)h23.x); l23.y = (__bf16)(f.w - (float)h23.y);
        uint2 hp, lp;
        hp.x = __builtin_bit_cast(unsigned, h01); hp.y = __builtin_bit_cast(unsigned, h23);
        lp.x = __builtin_bit_cast(unsigned, l01); lp.y = __builtin_bit_cast(unsigned, l23);
        *reinterpret_cast<uint2*>(hrow + j * 4) = hp;
        *reinterpret_cast<uint2*>(lrow + j * 4) = lp;
    }
    __syncthreads();

    int wave = tid >> 5, lane = tid & 31;
    int lm  = lane & 15;                          // A row-in-subtile / B column
    int sub = lane >> 4;                          // lane-half K interleave select
    int n0  = wave * 32;                          // column tiles n0, n0+16

    // B operand rows (col-major frag == contiguous K run): K base = sub*16
    const float* l1r0 = layer1 + (size_t)t * D * D + (size_t)(n0 + lm) * D + sub * 16;
    const float* l1r1 = l1r0 + 16 * D;
    const float* wr0  = W + (size_t)(n0 + lm) * D + sub * 16;
    const float* wr1  = wr0 + 16 * D;

    // A frag sources: per sub-tile s, row s*16+lm, K base = sub*16 (contiguous 16)
    const unsigned short* ahp[4];
    const unsigned short* alp[4];
    #pragma unroll
    for (int s = 0; s < 4; ++s) {
        int off = (s * 16 + lm) * ASTRIDE + sub * 16;
        ahp[s] = Ahi + off;
        alp[s] = Alo + off;
    }

    v8f accB[8], accJ[8];                         // [s*2 + ct]
    #pragma unroll
    for (int q = 0; q < 8; ++q) { accB[q] = (v8f){}; accJ[q] = (v8f){}; }

    for (int k = 0; k < D; k += 32) {
        // A frags (16x32 bf16 layout: elems 0-15 = K sub*16+e)
        v16bf Ah[4], Al[4];
        #pragma unroll
        for (int s = 0; s < 4; ++s) {
            v8bf a0 = *reinterpret_cast<const v8bf*>(ahp[s] + k);
            v8bf a1 = *reinterpret_cast<const v8bf*>(ahp[s] + k + 8);
            Ah[s] = __builtin_shufflevector(a0, a1, 0,1,2,3,4,5,6,7,8,9,10,11,12,13,14,15);
            v8bf b0 = *reinterpret_cast<const v8bf*>(alp[s] + k);
            v8bf b1 = *reinterpret_cast<const v8bf*>(alp[s] + k + 8);
            Al[s] = __builtin_shufflevector(b0, b1, 0,1,2,3,4,5,6,7,8,9,10,11,12,13,14,15);
        }
        #pragma unroll
        for (int ct = 0; ct < 2; ++ct) {
            v16bf Bh, Bl;
            split_frag((ct ? l1r1 : l1r0) + k, Bh, Bl);
            #pragma unroll
            for (int s = 0; s < 4; ++s) {
                v8f a = accB[s * 2 + ct];
                a = __builtin_amdgcn_wmma_f32_16x16x32_bf16(false, Ah[s], false, Bh, (short)0, a, false, false);
                a = __builtin_amdgcn_wmma_f32_16x16x32_bf16(false, Al[s], false, Bh, (short)0, a, false, false);
                a = __builtin_amdgcn_wmma_f32_16x16x32_bf16(false, Ah[s], false, Bl, (short)0, a, false, false);
                accB[s * 2 + ct] = a;
            }
            v16bf Wh, Wl;
            split_frag((ct ? wr1 : wr0) + k, Wh, Wl);
            #pragma unroll
            for (int s = 0; s < 4; ++s) {
                v8f a = accJ[s * 2 + ct];
                a = __builtin_amdgcn_wmma_f32_16x16x32_bf16(false, Ah[s], false, Wh, (short)0, a, false, false);
                a = __builtin_amdgcn_wmma_f32_16x16x32_bf16(false, Al[s], false, Wh, (short)0, a, false, false);
                a = __builtin_amdgcn_wmma_f32_16x16x32_bf16(false, Ah[s], false, Wl, (short)0, a, false, false);
                accJ[s * 2 + ct] = a;
            }
        }
    }

    // ---- epilogue: out = tanh(bmm) + adjust + b, scatter rows ----
    float b0 = bias[n0 + lm], b1 = bias[n0 + 16 + lm];
    int mb = sub * 8;                             // C/D layout: lanes 16-31 -> M+8
    #pragma unroll
    for (int s = 0; s < 4; ++s) {
        #pragma unroll
        for (int v = 0; v < 8; ++v) {
            int grow = rowIdx[s * 16 + mb + v];
            if (grow >= 0) {
                out[(size_t)grow * D + n0 + lm]      = tanhf(accB[s*2+0][v]) + accJ[s*2+0][v] + b0;
                out[(size_t)grow * D + n0 + 16 + lm] = tanhf(accB[s*2+1][v]) + accJ[s*2+1][v] + b1;
            }
        }
    }
}

extern "C" void kernel_launch(void* const* d_in, const int* in_sizes, int n_in,
                              void* d_out, int out_size, void* d_ws, size_t ws_size,
                              hipStream_t stream) {
    const int*   bi     = (const int*)d_in[0];
    const float* desc   = (const float*)d_in[1];
    const float* layer1 = (const float*)d_in[2];
    const float* W      = (const float*)d_in[3];
    const float* b      = (const float*)d_in[4];
    float* out = (float*)d_out;
    int*   ws  = (int*)d_ws;

    k_zero      <<<1, 64, 0, stream>>>(ws);
    k_count     <<<N_SAMP / 256, 256, 0, stream>>>(bi, ws);
    k_scan_tiles<<<1, 32, 0, stream>>>(ws);
    k_scatter   <<<N_SAMP / 256, 256, 0, stream>>>(bi, ws);
    // max tiles = sum ceil(cnt_t/64) <= 128 + 63 -> 192 blocks, excess exit early
    k_typed_gemm<<<192, 256, 0, stream>>>(desc, layer1, W, b, ws, out);
}
